// Attention_Layer_26044681683142
// MI455X (gfx1250) — matrix operand order repaired and checked by reading, not compiled, not run on hardware
//
#include <hip/hip_runtime.h>
#include <hip/hip_bf16.h>
#include <stdint.h>

#define D_IN  1024
#define N_SEQ 4096
#define B_SZ  4
#define N_CLS 4

typedef __attribute__((ext_vector_type(16))) __bf16 v16bf;
typedef __attribute__((ext_vector_type(8)))  float  v8f;
typedef __attribute__((ext_vector_type(4)))  unsigned int u32x4;
typedef __attribute__((ext_vector_type(8)))  int          i32x8;
typedef __attribute__((ext_vector_type(4)))  int          i32x4;

union FragBF {
  v16bf v;
  unsigned short h[16];
  uint4 q[2];
};

__device__ __forceinline__ unsigned short f32_to_bf16(float f) {
  union { float f; unsigned int u; } c; c.f = f;
  unsigned int u = c.u;
  u += 0x7FFFu + ((u >> 16) & 1u);      // round-to-nearest-even
  return (unsigned short)(u >> 16);
}

// ---------------------------------------------------------------------------
// 1a) W [k][n] f32  ->  WT [n][k] bf16   (contiguous-K B fragments)
// ---------------------------------------------------------------------------
__global__ void wt_bf16_kernel(const float* __restrict__ W,
                               unsigned short* __restrict__ WT) {
  int idx = blockIdx.x * blockDim.x + threadIdx.x;   // 0 .. 1024*1024-1
  int k = idx & (D_IN - 1);
  int n = idx >> 10;
  WT[(size_t)n * D_IN + k] = f32_to_bf16(W[(size_t)k * D_IN + n]);
}

// 1b) x f32 -> bf16 (same layout), done once instead of 3x inside the GEMM
__global__ void x_bf16_kernel(const float* __restrict__ X,
                              unsigned short* __restrict__ Xbf) {
  int idx = blockIdx.x * blockDim.x + threadIdx.x;   // one per 4 elements
  const float4 v = ((const float4*)X)[idx];
  ushort4 o;
  o.x = f32_to_bf16(v.x); o.y = f32_to_bf16(v.y);
  o.z = f32_to_bf16(v.z); o.w = f32_to_bf16(v.w);
  ((ushort4*)Xbf)[idx] = o;
}

// ---------------------------------------------------------------------------
// 2) Projection GEMM core: wave computes 16x64; all-bf16 fragment loads.
// ---------------------------------------------------------------------------
__device__ __forceinline__ void proj_tiles(const unsigned short* __restrict__ Xbf,
                                           const unsigned short* __restrict__ WT,
                                           int m0, int n0, int lane, v8f acc[4]) {
  const int arow = lane & 15;
  const int akb  = (lane >> 4) << 3;   // 0 or 8
  const int bcol = lane & 15;
  const int bkb  = (lane >> 4) << 3;   // 0 or 8
  const unsigned short* xrow = Xbf + (size_t)(m0 + arow) * D_IN;

  for (int k0 = 0; k0 < D_IN; k0 += 32) {
    FragBF a;
    a.q[0] = *(const uint4*)(xrow + k0 + akb);
    a.q[1] = *(const uint4*)(xrow + k0 + akb + 16);
#pragma unroll
    for (int t = 0; t < 4; ++t) {
      const unsigned short* wp =
          WT + (size_t)(n0 + t * 16 + bcol) * D_IN + k0 + bkb;
      FragBF bb;
      bb.q[0] = *(const uint4*)(wp);
      bb.q[1] = *(const uint4*)(wp + 16);
      acc[t] = __builtin_amdgcn_wmma_f32_16x16x32_bf16(
          false, a.v, false, bb.v, (short)0, acc[t], false, false);
    }
  }
}

// bf16 output variant (Q, K)
__global__ void proj_gemm_bf16_kernel(const unsigned short* __restrict__ Xbf,
                                      const unsigned short* __restrict__ WT,
                                      const float* __restrict__ bias,
                                      unsigned short* __restrict__ Y) {
  const int lane = threadIdx.x & 31;
  const int wave = threadIdx.x >> 5;
  const int m0 = blockIdx.x * 16;
  const int n0 = blockIdx.y * 512 + wave * 64;
  v8f acc[4] = {};
  proj_tiles(Xbf, WT, m0, n0, lane, acc);

  const int rbase = (lane >> 4) << 3;
#pragma unroll
  for (int t = 0; t < 4; ++t) {
    const int ncol = n0 + t * 16 + (lane & 15);
    const float bv = bias[ncol];
#pragma unroll
    for (int r = 0; r < 8; ++r)
      Y[(size_t)(m0 + rbase + r) * D_IN + ncol] = f32_to_bf16(acc[t][r] + bv);
  }
}

// f32 output variant (V)
__global__ void proj_gemm_f32_kernel(const unsigned short* __restrict__ Xbf,
                                     const unsigned short* __restrict__ WT,
                                     const float* __restrict__ bias,
                                     float* __restrict__ Y) {
  const int lane = threadIdx.x & 31;
  const int wave = threadIdx.x >> 5;
  const int m0 = blockIdx.x * 16;
  const int n0 = blockIdx.y * 512 + wave * 64;
  v8f acc[4] = {};
  proj_tiles(Xbf, WT, m0, n0, lane, acc);

  const int rbase = (lane >> 4) << 3;
#pragma unroll
  for (int t = 0; t < 4; ++t) {
    const int ncol = n0 + t * 16 + (lane & 15);
    const float bv = bias[ncol];
#pragma unroll
    for (int r = 0; r < 8; ++r)
      Y[(size_t)(m0 + rbase + r) * D_IN + ncol] = acc[t][r] + bv;
  }
}

// ---------------------------------------------------------------------------
// 3) Attention: one block per (b, 16-query strip). 512 threads = 16 waves.
//    Q strip staged via the Tensor Data Mover; scores live in LDS (256 KB).
// ---------------------------------------------------------------------------
__global__ void attn_kernel(const unsigned short* __restrict__ Qbf,
                            const unsigned short* __restrict__ Kbf,
                            float* __restrict__ wsum) {
  extern __shared__ char smem[];
  float* scores        = (float*)smem;                                    // [16][N_SEQ]
  unsigned short* qs   = (unsigned short*)(smem + 16 * N_SEQ * 4);        // [16][D_IN]
  float* invz          = (float*)(smem + 16 * N_SEQ * 4 + 16 * D_IN * 2); // [16]

  const int b     = blockIdx.x >> 8;       // N_SEQ/16 = 256 strips per batch
  const int strip = blockIdx.x & 255;
  const int tid   = threadIdx.x;
  const int lane  = tid & 31;
  const int wave  = tid >> 5;

  // --- TDM: DMA the contiguous 32KB Q strip into LDS (wave 0 only) ---
  if (wave == 0) {
    const unsigned long long gaddr =
        (unsigned long long)(uintptr_t)(Qbf + ((size_t)b * N_SEQ + strip * 16) * D_IN);
    const unsigned int lds_addr = (unsigned int)(uintptr_t)qs;  // LDS aperture low bits
    const unsigned int elems = 16 * D_IN;                        // 16384 bf16

    u32x4 g0;
    g0[0] = 1u;                                   // count=1, user descriptor
    g0[1] = lds_addr;                             // lds_addr [63:32]
    g0[2] = (unsigned int)(gaddr & 0xFFFFFFFFull);            // global_addr lo
    g0[3] = (unsigned int)((gaddr >> 32) & 0x1FFFFFFull) | (2u << 30); // hi + type=2

    i32x8 g1;
    g1[0] = (int)(1u << 16);                      // wg_mask=0, data_size=1 (2B)
    g1[1] = (int)(elems << 16);                   // tensor_dim0 lo16 in [63:48]
    g1[2] = (int)(1u << 16) | (int)(elems >> 16); // tensor_dim1=1, tensor_dim0 hi
    g1[3] = (int)(elems << 16);                   // tile_dim0 in [127:112]
    g1[4] = 1;                                    // tile_dim1=1, tile_dim2=0
    g1[5] = (int)elems;                           // tensor_dim0_stride lo32
    g1[6] = 0;                                    // stride hi, tensor_dim1_stride lo
    g1[7] = 0;

    i32x4 zero4 = {0, 0, 0, 0};
    i32x8 zero8 = {0, 0, 0, 0, 0, 0, 0, 0};
    __builtin_amdgcn_tensor_load_to_lds(g0, g1, zero4, zero4, zero8, 0);
    __builtin_amdgcn_s_wait_tensorcnt(0);
  }
  __syncthreads();

  const int arow = lane & 15;
  const int akb  = (lane >> 4) << 3;
  const int bcol = lane & 15;
  const int bkb  = (lane >> 4) << 3;
  const float scale = 0.03125f;            // 1/sqrt(1024)
  const unsigned short* qrow = qs + arow * D_IN;

  // --- score phase: S = Q K^T * scale ; 2 key tiles in flight per wave ---
#pragma unroll 1
  for (int j = 0; j < 8; ++j) {
    const int t0 = wave + 32 * j;            // tiles t0 and t0+16
    const int key0 = t0 * 16;
    const int key1 = (t0 + 16) * 16;
    const unsigned short* krow0 = Kbf + ((size_t)b * N_SEQ + key0 + bcol) * D_IN + bkb;
    const unsigned short* krow1 = Kbf + ((size_t)b * N_SEQ + key1 + bcol) * D_IN + bkb;
    __builtin_prefetch(krow0, 0, 1);
    __builtin_prefetch(krow1, 0, 1);
    v8f acc0 = {}, acc1 = {};
    for (int k0 = 0; k0 < D_IN; k0 += 32) {
      FragBF a, b0, b1;
      a.q[0]  = *(const uint4*)(qrow + k0 + akb);
      a.q[1]  = *(const uint4*)(qrow + k0 + akb + 16);
      b0.q[0] = *(const uint4*)(krow0 + k0);
      b0.q[1] = *(const uint4*)(krow0 + k0 + 16);
      b1.q[0] = *(const uint4*)(krow1 + k0);
      b1.q[1] = *(const uint4*)(krow1 + k0 + 16);
      acc0 = __builtin_amdgcn_wmma_f32_16x16x32_bf16(
          false, a.v, false, b0.v, (short)0, acc0, false, false);
      acc1 = __builtin_amdgcn_wmma_f32_16x16x32_bf16(
          false, a.v, false, b1.v, (short)0, acc1, false, false);
    }
    const int rbase = (lane >> 4) << 3;
#pragma unroll
    for (int r = 0; r < 8; ++r) {
      scores[(rbase + r) * N_SEQ + key0 + bcol] = acc0[r] * scale;
      scores[(rbase + r) * N_SEQ + key1 + bcol] = acc1[r] * scale;
    }
  }
  __syncthreads();

  // --- softmax per row: wave r owns row r ---
  {
    float* srow = scores + wave * N_SEQ;
    float m = -3.0e38f;
    for (int i = lane; i < N_SEQ; i += 32) m = fmaxf(m, srow[i]);
#pragma unroll
    for (int off = 16; off; off >>= 1) m = fmaxf(m, __shfl_xor(m, off, 32));
    float z = 0.f;
    for (int i = lane; i < N_SEQ; i += 32) {
      float e = __expf(srow[i] - m);
      srow[i] = e;
      z += e;
    }
#pragma unroll
    for (int off = 16; off; off >>= 1) z += __shfl_xor(z, off, 32);
    if (lane == 0) invz[wave] = 1.0f / z;
  }
  __syncthreads();

  // --- column sums of normalized probs -> global accumulate ---
  float iz[16];
#pragma unroll
  for (int r = 0; r < 16; ++r) iz[r] = invz[r];
  for (int m = tid; m < N_SEQ; m += blockDim.x) {
    float c = 0.f;
#pragma unroll
    for (int r = 0; r < 16; ++r) c += scores[r * N_SEQ + m] * iz[r];
    atomicAdd(&wsum[b * N_SEQ + m], c);
  }
}

// ---------------------------------------------------------------------------
// 4) pooled[b,d] = (1/N) * sum_m w[b,m] * V[b,m,d]
// ---------------------------------------------------------------------------
__global__ void pool_kernel(const float* __restrict__ Vf,
                            const float* __restrict__ wsum,
                            float* __restrict__ pooled) {
  const int b = blockIdx.y;
  const int d = blockIdx.x * blockDim.x + threadIdx.x;
  const float* vb = Vf + (size_t)b * N_SEQ * D_IN + d;
  const float* wb = wsum + b * N_SEQ;
  float acc = 0.f;
  for (int m = 0; m < N_SEQ; m += 4) {
    acc += wb[m + 0] * vb[(size_t)(m + 0) * D_IN];
    acc += wb[m + 1] * vb[(size_t)(m + 1) * D_IN];
    acc += wb[m + 2] * vb[(size_t)(m + 2) * D_IN];
    acc += wb[m + 3] * vb[(size_t)(m + 3) * D_IN];
  }
  pooled[b * D_IN + d] = acc * (1.0f / (float)N_SEQ);
}

// ---------------------------------------------------------------------------
// 5) logits = relu(pooled @ Wc + bc) ; one block per (b,c)
// ---------------------------------------------------------------------------
__global__ void head_kernel(const float* __restrict__ pooled,
                            const float* __restrict__ Wc,
                            const float* __restrict__ bc,
                            float* __restrict__ out) {
  const int b = blockIdx.x >> 2;
  const int c = blockIdx.x & 3;
  __shared__ float red[256];
  float acc = 0.f;
  for (int d = threadIdx.x; d < D_IN; d += 256)
    acc += pooled[b * D_IN + d] * Wc[d * N_CLS + c];
  red[threadIdx.x] = acc;
  __syncthreads();
  for (int s = 128; s; s >>= 1) {
    if (threadIdx.x < s) red[threadIdx.x] += red[threadIdx.x + s];
    __syncthreads();
  }
  if (threadIdx.x == 0) {
    float v = red[0] + bc[c];
    out[b * N_CLS + c] = v > 0.f ? v : 0.f;
  }
}

// ---------------------------------------------------------------------------
extern "C" void kernel_launch(void* const* d_in, const int* in_sizes, int n_in,
                              void* d_out, int out_size, void* d_ws, size_t ws_size,
                              hipStream_t stream) {
  const float* x  = (const float*)d_in[0];
  const float* Wk = (const float*)d_in[1];
  const float* bk = (const float*)d_in[2];
  const float* Wq = (const float*)d_in[3];
  const float* bq = (const float*)d_in[4];
  const float* Wv = (const float*)d_in[5];
  const float* bv = (const float*)d_in[6];
  const float* Wc = (const float*)d_in[7];
  const float* bc = (const float*)d_in[8];
  float* out = (float*)d_out;

  char* ws = (char*)d_ws;
  const size_t szWT  = (size_t)D_IN * D_IN * 2;            // 2 MB each
  const size_t szBF  = (size_t)B_SZ * N_SEQ * D_IN * 2;    // 32 MB each
  const size_t szVf  = (size_t)B_SZ * N_SEQ * D_IN * 4;    // 64 MB
  unsigned short* WkT = (unsigned short*)(ws);
  unsigned short* WqT = (unsigned short*)(ws + szWT);
  unsigned short* WvT = (unsigned short*)(ws + 2 * szWT);
  unsigned short* Xbf = (unsigned short*)(ws + 3 * szWT);
  unsigned short* Qbf = (unsigned short*)(ws + 3 * szWT + szBF);
  unsigned short* Kbf = (unsigned short*)(ws + 3 * szWT + 2 * szBF);
  float* Vf     = (float*)(ws + 3 * szWT + 3 * szBF);
  float* wsum   = (float*)(ws + 3 * szWT + 3 * szBF + szVf);
  float* pooled = (float*)(ws + 3 * szWT + 3 * szBF + szVf + (size_t)B_SZ * N_SEQ * 4);

  // 1) one-time conversions
  {
    dim3 g((D_IN * D_IN) / 256);
    wt_bf16_kernel<<<g, 256, 0, stream>>>(Wk, WkT);
    wt_bf16_kernel<<<g, 256, 0, stream>>>(Wq, WqT);
    wt_bf16_kernel<<<g, 256, 0, stream>>>(Wv, WvT);
    x_bf16_kernel<<<dim3((B_SZ * N_SEQ * D_IN / 4) / 256), 256, 0, stream>>>(x, Xbf);
  }

  // 2) projections (M = B*N = 16384)
  {
    dim3 g((B_SZ * N_SEQ) / 16, D_IN / 512);
    proj_gemm_bf16_kernel<<<g, 256, 0, stream>>>(Xbf, WqT, bq, Qbf);
    proj_gemm_bf16_kernel<<<g, 256, 0, stream>>>(Xbf, WkT, bk, Kbf);
    proj_gemm_f32_kernel <<<g, 256, 0, stream>>>(Xbf, WvT, bv, Vf);
  }

  // 3) attention probability column-sums
  (void)hipMemsetAsync(wsum, 0, (size_t)B_SZ * N_SEQ * 4, stream);
  {
    const size_t shmem = (size_t)16 * N_SEQ * 4 + (size_t)16 * D_IN * 2 + 64;
    attn_kernel<<<dim3(B_SZ * N_SEQ / 16), 512, shmem, stream>>>(Qbf, Kbf, wsum);
  }

  // 4) pooled = (1/N) w^T V
  pool_kernel<<<dim3(D_IN / 256, B_SZ), 256, 0, stream>>>(Vf, wsum, pooled);

  // 5) classifier head
  head_kernel<<<dim3(B_SZ * N_CLS), 256, 0, stream>>>(pooled, Wc, bc, out);
}
